// GCNModel_46883863003192
// MI455X (gfx1250) — matrix operand order repaired and checked by reading, not compiled, not run on hardware
//
#include <hip/hip_runtime.h>
#include <hip/hip_bf16.h>
#include <math.h>

#define HID 64

typedef __attribute__((ext_vector_type(16))) __bf16 v16bf;
typedef __attribute__((ext_vector_type(8)))  float  v8f;

// native f32 -> bf16 (backend emits v_cvt_*bf16_f32, often packed)
__device__ __forceinline__ __bf16 f2bf(float f) { return (__bf16)f; }

// hardware f32 atomic add (global_atomic_add_f32), relaxed, device scope
__device__ __forceinline__ void atomic_add_f32(float* p, float v) {
  __hip_atomic_fetch_add(p, v, __ATOMIC_RELAXED, __HIP_MEMORY_SCOPE_AGENT);
}

// Build a 16x32 bf16 A-fragment for one lane from two 8-float contiguous runs.
// Layout (ISA 7.12.2): lane = M + 16*half; e<8 -> K=8*half+e ; e>=8 -> K=16+8*half+(e-8)
__device__ __forceinline__ v16bf make_a_frag(const float* a0, const float* a1) {
  const float4 p0 = *(const float4*)a0;
  const float4 p1 = *(const float4*)(a0 + 4);
  const float4 p2 = *(const float4*)a1;
  const float4 p3 = *(const float4*)(a1 + 4);
  v16bf a;
  a[0]  = f2bf(p0.x); a[1]  = f2bf(p0.y); a[2]  = f2bf(p0.z); a[3]  = f2bf(p0.w);
  a[4]  = f2bf(p1.x); a[5]  = f2bf(p1.y); a[6]  = f2bf(p1.z); a[7]  = f2bf(p1.w);
  a[8]  = f2bf(p2.x); a[9]  = f2bf(p2.y); a[10] = f2bf(p2.z); a[11] = f2bf(p2.w);
  a[12] = f2bf(p3.x); a[13] = f2bf(p3.y); a[14] = f2bf(p3.z); a[15] = f2bf(p3.w);
  return a;
}

// ---------------- degree / symmetric norm ----------------
__global__ void k_deg_init(float* deg, int n) {
  int i = blockIdx.x * blockDim.x + threadIdx.x;
  if (i < n) deg[i] = 1.0f;               // self-loop contributes 1
}
__global__ void k_deg_scatter(const int* __restrict__ dst, float* __restrict__ deg, int E) {
  int i = blockIdx.x * blockDim.x + threadIdx.x;
  if (i < E) atomic_add_f32(&deg[dst[i]], 1.0f);
}
__global__ void k_dinv(float* deg, int n) {
  int i = blockIdx.x * blockDim.x + threadIdx.x;
  if (i < n) deg[i] = rsqrtf(deg[i]);     // deg >= 1 always
}

// ---------------- weight conversion ----------------
__global__ void k_f32_to_bf16(const float* __restrict__ in, __bf16* __restrict__ out, int n) {
  int i = blockIdx.x * blockDim.x + threadIdx.x;
  if (i < n) out[i] = f2bf(in[i]);
}

// ---------------- bf16 WMMA GEMM: C[M x 64] = A[M x K] (f32) * B[K x 64] (bf16) ----------------
// B (weights) staged once per block into LDS; one wave per 16-row strip;
// 4 accumulators cover N=64; K in steps of 32.
__global__ void k_gemm_bf16(const float* __restrict__ A, const __bf16* __restrict__ B,
                            float* __restrict__ C, int M, int K) {
  __shared__ __bf16 Bs[256 * HID];        // up to K=256 -> 32 KB of the WGP's 320 KB LDS

  const int tid = threadIdx.x;
  const int nElem = K * HID;
  for (int i = tid * 8; i < nElem; i += 256 * 8)          // 16B per thread per iter
    *(uint4*)(&Bs[i]) = *(const uint4*)(&B[i]);
  __syncthreads();

  const int lane = tid & 31;
  const int wave = tid >> 5;
  const int hf   = lane >> 4;
  const int l16  = lane & 15;
  const int rowBase = (blockIdx.x * 8 + wave) * 16;

  int m = rowBase + l16;
  if (m >= M) m = M - 1;          // clamp: keep EXEC all-ones for WMMA
  const float* arow = A + (size_t)m * K;

  v8f acc0 = {}, acc1 = {}, acc2 = {}, acc3 = {};
  for (int k0 = 0; k0 < K; k0 += 32) {
    v16bf a = make_a_frag(arow + k0 + 8 * hf, arow + k0 + 16 + 8 * hf);
    // B fragment: lane holds column n = l16 (+16*tile); element s -> K = 8*hf+s (s<8), 16+8*hf+(s-8) (s>=8)
    const __bf16* brow = &Bs[k0 * HID + l16];
    v16bf b0, b1, b2, b3;
#pragma unroll
    for (int s = 0; s < 16; ++s) {
      const __bf16* p = brow + ((s < 8) ? (8 * hf + s) : (16 + 8 * hf + (s - 8))) * HID;
      b0[s] = p[0]; b1[s] = p[16]; b2[s] = p[32]; b3[s] = p[48];
    }
    acc0 = __builtin_amdgcn_wmma_f32_16x16x32_bf16(false, a, false, b0, (short)0, acc0, false, false);
    acc1 = __builtin_amdgcn_wmma_f32_16x16x32_bf16(false, a, false, b1, (short)0, acc1, false, false);
    acc2 = __builtin_amdgcn_wmma_f32_16x16x32_bf16(false, a, false, b2, (short)0, acc2, false, false);
    acc3 = __builtin_amdgcn_wmma_f32_16x16x32_bf16(false, a, false, b3, (short)0, acc3, false, false);
  }
#pragma unroll
  for (int r = 0; r < 8; ++r) {
    int mm = rowBase + r + 8 * hf;
    if (mm < M) {
      float* crow = C + (size_t)mm * HID + l16;
      crow[0]  = acc0[r];
      crow[16] = acc1[r];
      crow[32] = acc2[r];
      crow[48] = acc3[r];
    }
  }
}

// ---------------- GCN aggregation ----------------
__global__ void k_self_init(const float* __restrict__ hx, const float* __restrict__ dinv,
                            float* __restrict__ hagg, int n) {
  int t = blockIdx.x * blockDim.x + threadIdx.x;
  if (t < n * HID) {
    float di = dinv[t >> 6];
    hagg[t] = hx[t] * di * di;   // self-loop term
  }
}
// 16 threads per edge, 4 columns each (float4 gather + 4 hardware f32 atomics)
__global__ void k_edge_scatter(const float* __restrict__ hx, const int* __restrict__ src,
                               const int* __restrict__ dst, const float* __restrict__ dinv,
                               float* __restrict__ hagg, int E) {
  long long t = (long long)blockIdx.x * blockDim.x + threadIdx.x;
  int e = (int)(t >> 4);
  if (e >= E) return;
  int c = ((int)t & 15) * 4;
  int s = src[e], d = dst[e];
  float w = dinv[s] * dinv[d];
  const float4 hv = *(const float4*)(hx + (size_t)s * HID + c);
  float* op = hagg + (size_t)d * HID + c;
  atomic_add_f32(op + 0, hv.x * w);
  atomic_add_f32(op + 1, hv.y * w);
  atomic_add_f32(op + 2, hv.z * w);
  atomic_add_f32(op + 3, hv.w * w);
}
__global__ void k_bias_act(float* __restrict__ h, const float* __restrict__ b, int n, int do_relu) {
  int t = blockIdx.x * blockDim.x + threadIdx.x;
  if (t < n * HID) {
    float v = h[t] + b[t & 63];
    if (do_relu) v = fmaxf(v, 0.0f);
    h[t] = v;
  }
}

// ---------------- fused edge decoder ----------------
// per wave: 16 edges. ef = [z[src], z[dst]] (K=128, bf16 in-register),
// e1 = relu(ef @ Wl1 + bl1) via 4 WMMAs x 4 k-steps, then dot(e1, Wl2) + bl2, sigmoid.
__global__ void k_decoder(const float* __restrict__ z, const int* __restrict__ src,
                          const int* __restrict__ dst, const __bf16* __restrict__ Wl1b,
                          const float* __restrict__ bl1, const float* __restrict__ Wl2,
                          const float* __restrict__ bl2, float* __restrict__ out, int E) {
  __shared__ __bf16 Ws[2 * HID * HID];    // Wl1 bf16: 16 KB, shared by all 8 waves
  const int tid = threadIdx.x;
  for (int i = tid * 8; i < 2 * HID * HID; i += 256 * 8)
    *(uint4*)(&Ws[i]) = *(const uint4*)(&Wl1b[i]);
  __syncthreads();

  const int lane = tid & 31;
  const int wave = tid >> 5;
  const int hf   = lane >> 4;
  const int l16  = lane & 15;
  const int eBase = (blockIdx.x * 8 + wave) * 16;

  int er = eBase + l16;
  if (er >= E) er = E - 1;       // clamp to keep EXEC all-ones
  const float* zs = z + (size_t)src[er] * HID;
  const float* zd = z + (size_t)dst[er] * HID;

  v8f acc[4];
#pragma unroll
  for (int t = 0; t < 4; ++t) { v8f zero = {}; acc[t] = zero; }

#pragma unroll
  for (int k0 = 0; k0 < 128; k0 += 32) {
    const float* ap = (k0 < 64) ? zs : zd;   // each 32-wide k-step lies fully in one half
    const int kb = k0 & 63;
    v16bf a = make_a_frag(ap + kb + 8 * hf, ap + kb + 16 + 8 * hf);
    const __bf16* brow = &Ws[k0 * HID + l16];
    v16bf b0, b1, b2, b3;
#pragma unroll
    for (int s = 0; s < 16; ++s) {
      const __bf16* p = brow + ((s < 8) ? (8 * hf + s) : (16 + 8 * hf + (s - 8))) * HID;
      b0[s] = p[0]; b1[s] = p[16]; b2[s] = p[32]; b3[s] = p[48];
    }
    acc[0] = __builtin_amdgcn_wmma_f32_16x16x32_bf16(false, a, false, b0, (short)0, acc[0], false, false);
    acc[1] = __builtin_amdgcn_wmma_f32_16x16x32_bf16(false, a, false, b1, (short)0, acc[1], false, false);
    acc[2] = __builtin_amdgcn_wmma_f32_16x16x32_bf16(false, a, false, b2, (short)0, acc[2], false, false);
    acc[3] = __builtin_amdgcn_wmma_f32_16x16x32_bf16(false, a, false, b3, (short)0, acc[3], false, false);
  }

  // bias + relu + dot with Wl2 along N (lane-local over 4 tiles, then 16-lane shuffle reduce)
  float part[8];
#pragma unroll
  for (int r = 0; r < 8; ++r) part[r] = 0.0f;
#pragma unroll
  for (int t = 0; t < 4; ++t) {
    const int n = 16 * t + l16;
    const float bb = bl1[n];
    const float w2 = Wl2[n];
#pragma unroll
    for (int r = 0; r < 8; ++r) {
      float v = acc[t][r] + bb;
      v = v > 0.0f ? v : 0.0f;
      part[r] += v * w2;
    }
  }
#pragma unroll
  for (int r = 0; r < 8; ++r) {
#pragma unroll
    for (int mask = 8; mask >= 1; mask >>= 1)
      part[r] += __shfl_xor(part[r], mask, 16);
  }
  if (l16 == 0) {
    const float bb2 = bl2[0];
#pragma unroll
    for (int r = 0; r < 8; ++r) {
      int ee = eBase + r + 8 * hf;
      if (ee < E) {
        float s = part[r] + bb2;
        out[ee] = 1.0f / (1.0f + __expf(-s));
      }
    }
  }
}

extern "C" void kernel_launch(void* const* d_in, const int* in_sizes, int n_in,
                              void* d_out, int out_size, void* d_ws, size_t ws_size,
                              hipStream_t stream) {
  (void)n_in; (void)out_size; (void)ws_size;
  const float* x   = (const float*)d_in[0];
  const int*   ei  = (const int*)d_in[1];
  const float* W1  = (const float*)d_in[2];
  const float* b1  = (const float*)d_in[3];
  const float* W2  = (const float*)d_in[4];
  const float* b2  = (const float*)d_in[5];
  const float* Wl1 = (const float*)d_in[6];
  const float* bl1 = (const float*)d_in[7];
  const float* Wl2 = (const float*)d_in[8];
  const float* bl2 = (const float*)d_in[9];

  const int IN_DIM = 256;
  const int N = in_sizes[0] / IN_DIM;
  const int E = in_sizes[1] / 2;
  const int* src = ei;
  const int* dst = ei + E;

  // ---- workspace carve (floats then bf16 weights) ----
  char* ws = (char*)d_ws;
  float* dinv = (float*)ws;  ws += (size_t)N * sizeof(float);
  float* hx   = (float*)ws;  ws += (size_t)N * HID * sizeof(float);   // pre-aggregation GEMM out (reused)
  float* hagg = (float*)ws;  ws += (size_t)N * HID * sizeof(float);   // conv1 output h1
  float* zbuf = (float*)ws;  ws += (size_t)N * HID * sizeof(float);   // conv2 output z
  __bf16* W1b  = (__bf16*)ws; ws += (size_t)IN_DIM * HID * sizeof(__bf16);
  __bf16* W2b  = (__bf16*)ws; ws += (size_t)HID * HID * sizeof(__bf16);
  __bf16* Wl1b = (__bf16*)ws; ws += (size_t)2 * HID * HID * sizeof(__bf16);

  const int T = 256;
  const unsigned gN  = (unsigned)((N + T - 1) / T);
  const unsigned gE  = (unsigned)((E + T - 1) / T);
  const unsigned gNH = (unsigned)(((size_t)N * HID + T - 1) / T);
  const unsigned gES = (unsigned)(((size_t)E * 16 + T - 1) / T);
  const unsigned gGemm = (unsigned)((N + 127) / 128);   // 8 waves x 16 rows per block
  const unsigned gDec  = (unsigned)((E + 127) / 128);

  // degrees + symmetric norm (dinv buffer doubles as deg)
  k_deg_init<<<gN, T, 0, stream>>>(dinv, N);
  k_deg_scatter<<<gE, T, 0, stream>>>(dst, dinv, E);
  k_dinv<<<gN, T, 0, stream>>>(dinv, N);

  // bf16 weights
  k_f32_to_bf16<<<(unsigned)((IN_DIM * HID + T - 1) / T), T, 0, stream>>>(W1, W1b, IN_DIM * HID);
  k_f32_to_bf16<<<(unsigned)((HID * HID + T - 1) / T), T, 0, stream>>>(W2, W2b, HID * HID);
  k_f32_to_bf16<<<(unsigned)((2 * HID * HID + T - 1) / T), T, 0, stream>>>(Wl1, Wl1b, 2 * HID * HID);

  // conv1: hx = x @ W1 ; aggregate ; + b1 ; relu
  k_gemm_bf16<<<gGemm, 256, 0, stream>>>(x, W1b, hx, N, IN_DIM);
  k_self_init<<<gNH, T, 0, stream>>>(hx, dinv, hagg, N);
  k_edge_scatter<<<gES, T, 0, stream>>>(hx, src, dst, dinv, hagg, E);
  k_bias_act<<<gNH, T, 0, stream>>>(hagg, b1, N, 1);

  // conv2: hx = h1 @ W2 ; aggregate ; + b2
  k_gemm_bf16<<<gGemm, 256, 0, stream>>>(hagg, W2b, hx, N, HID);
  k_self_init<<<gNH, T, 0, stream>>>(hx, dinv, zbuf, N);
  k_edge_scatter<<<gES, T, 0, stream>>>(hx, src, dst, dinv, zbuf, E);
  k_bias_act<<<gNH, T, 0, stream>>>(zbuf, b2, N, 0);

  // fused decoder: gather -> WMMA edge MLP -> relu -> dot Wl2 -> sigmoid
  k_decoder<<<gDec, 256, 0, stream>>>(zbuf, src, dst, Wl1b, bl1, Wl2, bl2, (float*)d_out, E);
}
